// QKVAttention_9680856285882
// MI455X (gfx1250) — matrix operand order repaired and checked by reading, not compiled, not run on hardware
//
#include <hip/hip_runtime.h>

typedef __bf16 bf16_t;
typedef bf16_t v16bf __attribute__((ext_vector_type(16)));
typedef bf16_t v8bf  __attribute__((ext_vector_type(8)));
typedef float  v8f   __attribute__((ext_vector_type(8)));

#define T_SEQ  2048
#define C_HEAD 64
#define NCH    192    // 3 * C_HEAD

#define SKEYS  64     // keys per chunk staged in LDS
#define LROW   72     // padded LDS row stride (bf16 elems): 144B rows -> conflict-free

static __device__ __forceinline__ v8f wmma_bf16(v16bf a, v16bf b, v8f c) {
  // D = A(16x32 bf16) x B(32x16 bf16) + C(16x16 f32)
  return __builtin_amdgcn_wmma_f32_16x16x32_bf16(false, a, false, b, (short)0, c,
                                                 false, false);
}

// A-fragment gather from LDS: two contiguous 8-elem (16B) runs -> 2x ds_load_b128
static __device__ __forceinline__ v16bf lds_frag(const bf16_t* base) {
  const v8bf lo = *(const v8bf*)base;          // K = h*16 + 0..7
  const v8bf hi = *(const v8bf*)(base + 8);    // K = h*16 + 8..15
  v16bf r;
  #pragma unroll
  for (int j = 0; j < 8; ++j) { r[j] = lo[j]; r[8 + j] = hi[j]; }
  return r;
}

__global__ __launch_bounds__(128)
void qkv_attn_fa_bf16(const float* __restrict__ qkv, float* __restrict__ out)
{
  const int lane = threadIdx.x & 31;
  const int wave = threadIdx.x >> 5;
  const int n    = lane & 15;   // tile column (query) / A-fragment row
  const int h    = lane >> 4;   // half-wave select
  const int m    = n;           // A-fragment row

  const int b  = blockIdx.y;
  const int t0 = (blockIdx.x * 4 + wave) * 32;   // this wave's 32 queries (2 tiles)

  const float* __restrict__ qp = qkv + (size_t)b * NCH * T_SEQ;
  const float* __restrict__ kp = qp + (size_t)C_HEAD * T_SEQ;
  const float* __restrict__ vp = qp + (size_t)(2 * C_HEAD) * T_SEQ;

  __shared__ __align__(16) bf16_t ldsK[SKEYS * LROW];   // [s][c] bf16, padded
  __shared__ __align__(16) bf16_t ldsV[C_HEAD * LROW];  // [c][s] bf16, padded

  // 1/sqrt(C) softmax scale folded with log2(e) so the hw base-2 exp applies directly
  const float qscale = 0.125f * 1.44269504088896341f;

  // ---- Q as B-fragments (K-dim = channels): lane(h,n) element e -> c = ch*32 + h*16 + e
  v16bf bq[2][2];   // [query tile][channel chunk]
  #pragma unroll
  for (int qi = 0; qi < 2; ++qi) {
    const int tq = t0 + qi * 16 + n;
    #pragma unroll
    for (int ch = 0; ch < 2; ++ch) {
      #pragma unroll
      for (int e = 0; e < 16; ++e) {
        const int c = ch * 32 + h * 16 + e;
        bq[qi][ch][e] = (bf16_t)(qp[(size_t)c * T_SEQ + tq] * qscale);
      }
    }
  }

  // O^T accumulators: per query tile, 4 tiles of (16 channels x 16 queries) f32
  v8f o[2][4];
  #pragma unroll
  for (int qi = 0; qi < 2; ++qi)
    #pragma unroll
    for (int cc = 0; cc < 4; ++cc)
      #pragma unroll
      for (int r = 0; r < 8; ++r) o[qi][cc][r] = 0.f;

  float m_run[2] = {-3.0e38f, -3.0e38f};
  float l_run[2] = {0.f, 0.f};

  for (int s0 = 0; s0 < T_SEQ; s0 += SKEYS) {
    // ============== cooperative staging: 64-key K/V chunk -> LDS as bf16 ==========
    {
      // K: global [c][s] -> LDS [s][c] (transpose). thread -> (s, 32-channel group)
      const int ks = threadIdx.x & 63;
      const int cg = threadIdx.x >> 6;           // 0..1
      float f[32];
      #pragma unroll
      for (int e = 0; e < 32; ++e)
        f[e] = kp[(size_t)(cg * 32 + e) * T_SEQ + s0 + ks];
      #pragma unroll
      for (int g = 0; g < 4; ++g) {
        v8bf p;
        #pragma unroll
        for (int j = 0; j < 8; ++j) p[j] = (bf16_t)f[g * 8 + j];
        *(v8bf*)&ldsK[ks * LROW + cg * 32 + g * 8] = p;
      }

      // V: global [c][s] -> LDS [c][s]. thread -> (c, 32-key half)
      const int vc = threadIdx.x & 63;
      const int sh = threadIdx.x >> 6;           // 0..1
      const float4* v4 =
          reinterpret_cast<const float4*>(vp + (size_t)vc * T_SEQ + s0 + sh * 32);
      #pragma unroll
      for (int g = 0; g < 4; ++g) {
        const float4 a0 = v4[2 * g];
        const float4 a1 = v4[2 * g + 1];
        v8bf p;
        p[0] = (bf16_t)a0.x; p[1] = (bf16_t)a0.y; p[2] = (bf16_t)a0.z; p[3] = (bf16_t)a0.w;
        p[4] = (bf16_t)a1.x; p[5] = (bf16_t)a1.y; p[6] = (bf16_t)a1.z; p[7] = (bf16_t)a1.w;
        *(v8bf*)&ldsV[vc * LROW + sh * 32 + g * 8] = p;
      }
    }
    if (s0 + SKEYS < T_SEQ) {  // speculative prefetch of next chunk lines
      __builtin_prefetch(kp + (size_t)(threadIdx.x >> 1) * T_SEQ + s0 + SKEYS, 0, 1);
      __builtin_prefetch(vp + (size_t)(threadIdx.x >> 1) * T_SEQ + s0 + SKEYS, 0, 1);
    }
    __syncthreads();

    // ---- GEMM1 + online softmax per query tile; produce P^T B-fragments
    v16bf b2[2][2];   // [query tile][32-key sub-chunk]
    #pragma unroll
    for (int qi = 0; qi < 2; ++qi) {
      // S^T tiles for 4 key sub-tiles, accumulated over 64 channels
      v8f d1[4];
      #pragma unroll
      for (int st = 0; st < 4; ++st) {
        const bf16_t* krow = &ldsK[(st * 16 + m) * LROW + h * 16];
        const v16bf ka0 = lds_frag(krow);        // c-chunk 0
        const v16bf ka1 = lds_frag(krow + 32);   // c-chunk 1
        v8f z;
        #pragma unroll
        for (int r = 0; r < 8; ++r) z[r] = 0.f;
        d1[st] = wmma_bf16(ka0, bq[qi][0], z);
        d1[st] = wmma_bf16(ka1, bq[qi][1], d1[st]);
      }

      // row max over all 64 keys of this chunk (per lane = per query column)
      float mx = d1[0][0];
      #pragma unroll
      for (int st = 0; st < 4; ++st)
        #pragma unroll
        for (int r = 0; r < 8; ++r) mx = fmaxf(mx, d1[st][r]);
      mx = fmaxf(mx, __shfl_xor(mx, 16, 32));

      const float m_new = fmaxf(m_run[qi], mx);
      const float corr  = exp2f(m_run[qi] - m_new);
      m_run[qi] = m_new;

      float p[4][8];
      float s_loc = 0.f;
      #pragma unroll
      for (int st = 0; st < 4; ++st)
        #pragma unroll
        for (int r = 0; r < 8; ++r) {
          p[st][r] = exp2f(d1[st][r] - m_new);
          s_loc += p[st][r];
        }
      s_loc += __shfl_xor(s_loc, 16, 32);
      l_run[qi] = l_run[qi] * corr + s_loc;

      #pragma unroll
      for (int cc = 0; cc < 4; ++cc)
        #pragma unroll
        for (int r = 0; r < 8; ++r) o[qi][cc][r] *= corr;

      // relayout P^T (D-layout) -> B-fragments via cross-half shuffles.
      // Sub-chunk g covers keys g*32..g*32+31; B-frag lane(h,n) elem e = key h*16+e.
      #pragma unroll
      for (int g = 0; g < 2; ++g) {
        #pragma unroll
        for (int r = 0; r < 8; ++r) {
          const float pao = __shfl_xor(p[2 * g][r],     16, 32);
          const float pbo = __shfl_xor(p[2 * g + 1][r], 16, 32);
          const float lo = h ? pbo           : p[2 * g][r];   // rows g*32 + h*16 + 0..7
          const float hi = h ? p[2 * g + 1][r] : pao;         // rows g*32 + h*16 + 8..15
          b2[qi][g][r]     = (bf16_t)lo;
          b2[qi][g][8 + r] = (bf16_t)hi;
        }
      }
    }

    // ---- GEMM2: O^T += V * P^T ; each V fragment feeds both query tiles
    #pragma unroll
    for (int cc = 0; cc < 4; ++cc) {
      const bf16_t* vrow = &ldsV[(cc * 16 + m) * LROW + h * 16];
      #pragma unroll
      for (int g = 0; g < 2; ++g) {
        const v16bf av = lds_frag(vrow + g * 32);   // s = g*32 + (A-layout K pattern)
        #pragma unroll
        for (int qi = 0; qi < 2; ++qi)
          o[qi][cc] = wmma_bf16(av, b2[qi][g], o[qi][cc]);
      }
    }

    __syncthreads();   // protect LDS before next chunk overwrites it
  }

  // ---- epilogue: normalize and store O^T -> out[b, c, t]
  float* __restrict__ op = out + (size_t)b * C_HEAD * T_SEQ;
  #pragma unroll
  for (int qi = 0; qi < 2; ++qi) {
    const float inv_l = 1.0f / l_run[qi];
    const int tq = t0 + qi * 16 + n;
    #pragma unroll
    for (int cc = 0; cc < 4; ++cc) {
      #pragma unroll
      for (int r = 0; r < 8; ++r) {
        const int c = cc * 16 + h * 8 + r;
        op[(size_t)c * T_SEQ + tq] = o[qi][cc][r] * inv_l;
      }
    }
  }
}

extern "C" void kernel_launch(void* const* d_in, const int* in_sizes, int n_in,
                              void* d_out, int out_size, void* d_ws, size_t ws_size,
                              hipStream_t stream) {
  (void)n_in; (void)d_ws; (void)ws_size; (void)out_size;
  const float* qkv = (const float*)d_in[0];
  float* out = (float*)d_out;
  const int B = in_sizes[0] / (NCH * T_SEQ);   // 32
  dim3 grid(T_SEQ / 128, B);                   // 4 waves/block * 32 queries/wave
  dim3 block(128);
  hipLaunchKernelGGL(qkv_attn_fa_bf16, grid, block, 0, stream, qkv, out);
}
